// EntropyInvarianceAttention_22101901705275
// MI455X (gfx1250) — matrix operand order repaired and checked by reading, not compiled, not run on hardware
//
#include <hip/hip_runtime.h>
#include <hip/hip_bf16.h>

#define NH   8
#define HD   64
#define LQK  2048
#define QT   128    // queries per block (8 waves * 16)
#define KC   64     // keys per chunk
#define BH   32     // B * NH

typedef __attribute__((ext_vector_type(16))) _Float16 v16h;
typedef __attribute__((ext_vector_type(8)))  _Float16 v8h;
typedef __attribute__((ext_vector_type(4)))  _Float16 v4h;
typedef __attribute__((ext_vector_type(8)))  float    v8f;

union F16x16 { v16h v; v8h h2[2]; _Float16 h[16]; };

// ---------------------------------------------------------------------------
// Pass 1: convert K -> f16 transposed [bh][key][d], V -> f16 direct [bh][d][key]
// ---------------------------------------------------------------------------
__global__ __launch_bounds__(256)
void eia_cvt_kv(const float* __restrict__ K, const float* __restrict__ V,
                _Float16* __restrict__ Kh, _Float16* __restrict__ Vh)
{
  __shared__ float Ts[64][65];            // padded to dodge bank conflicts
  const int bh  = blockIdx.y;
  const int kb  = blockIdx.x * 64;
  const int tid = threadIdx.x;
  const float* Kg = K + (size_t)bh * HD * LQK;
  const float* Vg = V + (size_t)bh * HD * LQK;

  // K tile 64d x 64key -> LDS (f32)
  #pragma unroll
  for (int i = 0; i < 4; ++i) {
    int s  = tid + i * 256;
    int d  = s >> 4;
    int kq = (s & 15) << 2;
    float4 val = *(const float4*)(Kg + (size_t)d * LQK + kb + kq);
    Ts[d][kq + 0] = val.x; Ts[d][kq + 1] = val.y;
    Ts[d][kq + 2] = val.z; Ts[d][kq + 3] = val.w;
  }
  __syncthreads();

  // transposed f16 write: Kh[key][d], coalesced along d (32B per thread)
  {
    int key = tid >> 2;
    int dc  = (tid & 3) << 4;
    F16x16 t;
    #pragma unroll
    for (int j = 0; j < 16; ++j) t.h[j] = (_Float16)Ts[dc + j][key];
    _Float16* dst = Kh + (size_t)bh * LQK * HD + (size_t)(kb + key) * HD + dc;
    *(v8h*)dst       = t.h2[0];
    *(v8h*)(dst + 8) = t.h2[1];
  }

  // V elementwise convert: Vh[d][key]
  #pragma unroll
  for (int i = 0; i < 2; ++i) {
    int s  = tid + i * 256;               // 512 slots of 8 floats
    int d  = s >> 3;
    int kq = (s & 7) << 3;
    float4 a  = *(const float4*)(Vg + (size_t)d * LQK + kb + kq);
    float4 b2 = *(const float4*)(Vg + (size_t)d * LQK + kb + kq + 4);
    v8h p = { (_Float16)a.x,  (_Float16)a.y,  (_Float16)a.z,  (_Float16)a.w,
              (_Float16)b2.x, (_Float16)b2.y, (_Float16)b2.z, (_Float16)b2.w };
    *(v8h*)(Vh + (size_t)bh * HD * LQK + (size_t)d * LQK + kb + kq) = p;
  }
}

// ---------------------------------------------------------------------------
// Pass 2: flash attention, K/V chunks staged via async global->LDS copies
// (double-buffered, ASYNCcnt pipelined)
// ---------------------------------------------------------------------------
__global__ __launch_bounds__(256)
void eia_fwd_async(const float* __restrict__ Q, const _Float16* __restrict__ Kh,
                   const _Float16* __restrict__ Vh, const int* __restrict__ KL,
                   float* __restrict__ O)
{
  __shared__ __align__(16) _Float16 Qs[QT][HD];      // [q][d] scaled f16, 16 KB
  __shared__ __align__(16) _Float16 Ks[2][KC][HD];   // [key][d] f16, 2 x 8 KB
  __shared__ __align__(16) _Float16 Vs[2][HD][KC];   // [d][key] f16, 2 x 8 KB

  const int bh   = blockIdx.y;
  const int b    = bh >> 3;
  const int q0   = blockIdx.x * QT;
  const int tid  = threadIdx.x;
  const int lane = tid & 31;
  const int ln   = lane & 15;
  const int hsel = lane >> 4;
  const int wv   = tid >> 5;

  const float scale = (1.0f / (8.0f * __logf(20.0f))) * __logf((float)KL[b]);

  const float*    Qg = Q  + (size_t)bh * HD * LQK;
  const _Float16* Kb = Kh + (size_t)bh * LQK * HD;   // [key][d]
  const _Float16* Vb = Vh + (size_t)bh * HD * LQK;   // [d][key]

  // issue one chunk of async copies: 4 x b128 per thread (2 K + 2 V)
  auto issue_chunk = [&](int buf, int kb) {
    const unsigned ldsK = (unsigned)(size_t)&Ks[buf][0][0];
    const unsigned ldsV = (unsigned)(size_t)&Vs[buf][0][0];
    const _Float16* ksrc = Kb + (size_t)kb * HD;     // 8 KB contiguous chunk
    const _Float16* vsrc = Vb + kb;                  // 64 rows x 128 B
    #pragma unroll
    for (int i = 0; i < 2; ++i) {
      unsigned off = (unsigned)tid * 16u + (unsigned)i * 4096u;
      asm volatile("global_load_async_to_lds_b128 %0, %1, %2"
                   :: "v"(ldsK + off), "v"(off), "s"(ksrc) : "memory");
    }
    #pragma unroll
    for (int i = 0; i < 2; ++i) {
      unsigned d  = (unsigned)(tid >> 3) + (unsigned)i * 32u;
      unsigned go = d * (LQK * 2u) + (unsigned)(tid & 7) * 16u;  // global bytes
      unsigned lo = d * (KC * 2u)  + (unsigned)(tid & 7) * 16u;  // LDS bytes
      asm volatile("global_load_async_to_lds_b128 %0, %1, %2"
                   :: "v"(ldsV + lo), "v"(go), "s"(vsrc) : "memory");
    }
  };

  // ---- stage scaled Q tile, transposed to [q][d] f16 (once per block) ----
  #pragma unroll
  for (int i = 0; i < 8; ++i) {
    int s  = tid + i * 256;
    int d  = s >> 5;
    int qq = (s & 31) << 2;
    float4 val = *(const float4*)(Qg + (size_t)d * LQK + q0 + qq);
    Qs[qq + 0][d] = (_Float16)(val.x * scale);
    Qs[qq + 1][d] = (_Float16)(val.y * scale);
    Qs[qq + 2][d] = (_Float16)(val.z * scale);
    Qs[qq + 3][d] = (_Float16)(val.w * scale);
  }
  issue_chunk(0, 0);              // prime the async pipeline
  __syncthreads();

  // ---- per-wave Q B-fragments (K=d sequential halves, N=q=lane%16) ----
  const int qi = wv * 16 + ln;
  v16h qf0, qf1;
  {
    F16x16 t;
    t.h2[0] = *(const v8h*)(&Qs[qi][hsel * 16 + 0]);
    t.h2[1] = *(const v8h*)(&Qs[qi][hsel * 16 + 8]);
    qf0 = t.v;
    t.h2[0] = *(const v8h*)(&Qs[qi][32 + hsel * 16 + 0]);
    t.h2[1] = *(const v8h*)(&Qs[qi][32 + hsel * 16 + 8]);
    qf1 = t.v;
  }

  v8f o0 = {}, o1 = {}, o2 = {}, o3 = {};
  float m = -1e30f, l = 0.0f;

  for (int kb = 0; kb < LQK; kb += KC) {
    const int cur = (kb >> 6) & 1;
    if (kb + KC < LQK) {
      issue_chunk(cur ^ 1, kb + KC);                 // next chunk in flight
      asm volatile("s_wait_asynccnt 0x4" ::: "memory");  // current chunk landed
    } else {
      asm volatile("s_wait_asynccnt 0x0" ::: "memory");
    }
    __syncthreads();

    // ---- S^T = K^T * Q  (M=key 4x16, N=q, K=d 2x32) ----
    v8f st[4];
    #pragma unroll
    for (int t = 0; t < 4; ++t) {
      v8f acc = {};
      #pragma unroll
      for (int kk = 0; kk < 2; ++kk) {
        F16x16 af;
        const int row  = t * 16 + ln;
        const int dofs = kk * 32 + hsel * 16;
        af.h2[0] = *(const v8h*)(&Ks[cur][row][dofs]);
        af.h2[1] = *(const v8h*)(&Ks[cur][row][dofs + 8]);
        acc = __builtin_amdgcn_wmma_f32_16x16x32_f16(
            false, af.v, false, (kk ? qf1 : qf0), (short)0, acc, false, false);
      }
      st[t] = acc;
    }

    // ---- online softmax (per-lane query column) ----
    float cm = -1e30f;
    #pragma unroll
    for (int t = 0; t < 4; ++t)
      #pragma unroll
      for (int j = 0; j < 8; ++j)
        cm = fmaxf(cm, st[t][j]);
    cm = fmaxf(cm, __shfl_xor(cm, 16, 32));
    const float mn    = fmaxf(m, cm);
    const float alpha = __expf(m - mn);
    m = mn;
    float rs = 0.0f;
    #pragma unroll
    for (int t = 0; t < 4; ++t)
      #pragma unroll
      for (int j = 0; j < 8; ++j) {
        float p = __expf(st[t][j] - mn);
        st[t][j] = p;
        rs += p;
      }
    rs += __shfl_xor(rs, 16, 32);
    l = l * alpha + rs;
    #pragma unroll
    for (int j = 0; j < 8; ++j) { o0[j] *= alpha; o1[j] *= alpha; o2[j] *= alpha; o3[j] *= alpha; }

    // ---- O += V * P^T  (M=d 4x16, N=q, K=keys 2x32) ----
    #pragma unroll
    for (int kk = 0; kk < 2; ++kk) {
      F16x16 bf;
      v8f pa = st[kk * 2], pb = st[kk * 2 + 1];
      #pragma unroll
      for (int j = 0; j < 8; ++j) {
        float xa = __shfl_xor(pa[j], 16, 32);
        float xb = __shfl_xor(pb[j], 16, 32);
        bf.h[j]     = (_Float16)(hsel ? xb    : pa[j]);
        bf.h[j + 8] = (_Float16)(hsel ? pb[j] : xa);
      }
      #pragma unroll
      for (int n = 0; n < 4; ++n) {
        F16x16 vf;
        const int drow = n * 16 + ln;
        const int kofs = kk * 32 + hsel * 16;
        vf.h2[0] = *(const v8h*)(&Vs[cur][drow][kofs]);
        vf.h2[1] = *(const v8h*)(&Vs[cur][drow][kofs + 8]);
        v8f& oc = (n == 0 ? o0 : n == 1 ? o1 : n == 2 ? o2 : o3);
        oc = __builtin_amdgcn_wmma_f32_16x16x32_f16(
            false, vf.v, false, bf.v, (short)0, oc, false, false);
      }
    }
    __syncthreads();   // all reads of buf `cur` done before it is refilled
  }

  const float inv = 1.0f / l;
  float* Og = O + (size_t)bh * HD * LQK + q0 + wv * 16;
  #pragma unroll
  for (int n = 0; n < 4; ++n) {
    v8f oc = (n == 0 ? o0 : n == 1 ? o1 : n == 2 ? o2 : o3);
    #pragma unroll
    for (int j = 0; j < 8; ++j)
      Og[(size_t)(n * 16 + hsel * 8 + j) * LQK + ln] = oc[j] * inv;
  }
}

// ---------------------------------------------------------------------------
// Fallback: fully fused single-pass kernel (no workspace needed)
// ---------------------------------------------------------------------------
__global__ __launch_bounds__(256)
void eia_fwd_fused(const float* __restrict__ Q, const float* __restrict__ K,
                   const float* __restrict__ V, const int* __restrict__ KL,
                   float* __restrict__ O)
{
  __shared__ __align__(16) _Float16 Qs[QT][HD];
  __shared__ __align__(16) _Float16 Ks[KC][HD];
  __shared__ __align__(16) _Float16 Vs[HD][KC];

  const int bh   = blockIdx.y;
  const int b    = bh >> 3;
  const int q0   = blockIdx.x * QT;
  const int tid  = threadIdx.x;
  const int lane = tid & 31;
  const int ln   = lane & 15;
  const int hsel = lane >> 4;
  const int wv   = tid >> 5;

  const float scale = (1.0f / (8.0f * __logf(20.0f))) * __logf((float)KL[b]);

  const float* Qg = Q + (size_t)bh * HD * LQK;
  const float* Kg = K + (size_t)bh * HD * LQK;
  const float* Vg = V + (size_t)bh * HD * LQK;

  #pragma unroll
  for (int i = 0; i < 8; ++i) {
    int s  = tid + i * 256;
    int d  = s >> 5;
    int qq = (s & 31) << 2;
    float4 val = *(const float4*)(Qg + (size_t)d * LQK + q0 + qq);
    Qs[qq + 0][d] = (_Float16)(val.x * scale);
    Qs[qq + 1][d] = (_Float16)(val.y * scale);
    Qs[qq + 2][d] = (_Float16)(val.z * scale);
    Qs[qq + 3][d] = (_Float16)(val.w * scale);
  }
  __syncthreads();

  const int qi = wv * 16 + ln;
  v16h qf0, qf1;
  {
    F16x16 t;
    t.h2[0] = *(const v8h*)(&Qs[qi][hsel * 16 + 0]);
    t.h2[1] = *(const v8h*)(&Qs[qi][hsel * 16 + 8]);
    qf0 = t.v;
    t.h2[0] = *(const v8h*)(&Qs[qi][32 + hsel * 16 + 0]);
    t.h2[1] = *(const v8h*)(&Qs[qi][32 + hsel * 16 + 8]);
    qf1 = t.v;
  }

  v8f o0 = {}, o1 = {}, o2 = {}, o3 = {};
  float m = -1e30f, l = 0.0f;

  for (int kb = 0; kb < LQK; kb += KC) {
    __syncthreads();
    #pragma unroll
    for (int i = 0; i < 4; ++i) {
      int s  = tid + i * 256;
      int d  = s >> 4;
      int kq = (s & 15) << 2;
      float4 kv = *(const float4*)(Kg + (size_t)d * LQK + kb + kq);
      Ks[kq + 0][d] = (_Float16)kv.x;
      Ks[kq + 1][d] = (_Float16)kv.y;
      Ks[kq + 2][d] = (_Float16)kv.z;
      Ks[kq + 3][d] = (_Float16)kv.w;
      float4 vv = *(const float4*)(Vg + (size_t)d * LQK + kb + kq);
      v4h pv = { (_Float16)vv.x, (_Float16)vv.y, (_Float16)vv.z, (_Float16)vv.w };
      *(v4h*)(&Vs[d][kq]) = pv;
    }
    __syncthreads();

    v8f st[4];
    #pragma unroll
    for (int t = 0; t < 4; ++t) {
      v8f acc = {};
      #pragma unroll
      for (int kk = 0; kk < 2; ++kk) {
        F16x16 af;
        const int row  = t * 16 + ln;
        const int dofs = kk * 32 + hsel * 16;
        af.h2[0] = *(const v8h*)(&Ks[row][dofs]);
        af.h2[1] = *(const v8h*)(&Ks[row][dofs + 8]);
        acc = __builtin_amdgcn_wmma_f32_16x16x32_f16(
            false, af.v, false, (kk ? qf1 : qf0), (short)0, acc, false, false);
      }
      st[t] = acc;
    }

    float cm = -1e30f;
    #pragma unroll
    for (int t = 0; t < 4; ++t)
      #pragma unroll
      for (int j = 0; j < 8; ++j)
        cm = fmaxf(cm, st[t][j]);
    cm = fmaxf(cm, __shfl_xor(cm, 16, 32));
    const float mn    = fmaxf(m, cm);
    const float alpha = __expf(m - mn);
    m = mn;
    float rs = 0.0f;
    #pragma unroll
    for (int t = 0; t < 4; ++t)
      #pragma unroll
      for (int j = 0; j < 8; ++j) {
        float p = __expf(st[t][j] - mn);
        st[t][j] = p;
        rs += p;
      }
    rs += __shfl_xor(rs, 16, 32);
    l = l * alpha + rs;
    #pragma unroll
    for (int j = 0; j < 8; ++j) { o0[j] *= alpha; o1[j] *= alpha; o2[j] *= alpha; o3[j] *= alpha; }

    #pragma unroll
    for (int kk = 0; kk < 2; ++kk) {
      F16x16 bf;
      v8f pa = st[kk * 2], pb = st[kk * 2 + 1];
      #pragma unroll
      for (int j = 0; j < 8; ++j) {
        float xa = __shfl_xor(pa[j], 16, 32);
        float xb = __shfl_xor(pb[j], 16, 32);
        bf.h[j]     = (_Float16)(hsel ? xb    : pa[j]);
        bf.h[j + 8] = (_Float16)(hsel ? pb[j] : xa);
      }
      #pragma unroll
      for (int n = 0; n < 4; ++n) {
        F16x16 vf;
        const int drow = n * 16 + ln;
        const int kofs = kk * 32 + hsel * 16;
        vf.h2[0] = *(const v8h*)(&Vs[drow][kofs]);
        vf.h2[1] = *(const v8h*)(&Vs[drow][kofs + 8]);
        v8f& oc = (n == 0 ? o0 : n == 1 ? o1 : n == 2 ? o2 : o3);
        oc = __builtin_amdgcn_wmma_f32_16x16x32_f16(
            false, vf.v, false, bf.v, (short)0, oc, false, false);
      }
    }
  }

  const float inv = 1.0f / l;
  float* Og = O + (size_t)bh * HD * LQK + q0 + wv * 16;
  #pragma unroll
  for (int n = 0; n < 4; ++n) {
    v8f oc = (n == 0 ? o0 : n == 1 ? o1 : n == 2 ? o2 : o3);
    #pragma unroll
    for (int j = 0; j < 8; ++j)
      Og[(size_t)(n * 16 + hsel * 8 + j) * LQK + ln] = oc[j] * inv;
  }
}

extern "C" void kernel_launch(void* const* d_in, const int* in_sizes, int n_in,
                              void* d_out, int out_size, void* d_ws, size_t ws_size,
                              hipStream_t stream) {
  const float* q  = (const float*)d_in[0];
  const float* k  = (const float*)d_in[1];
  const float* v  = (const float*)d_in[2];
  const int*   kl = (const int*)d_in[3];
  float* out = (float*)d_out;

  const size_t elems = (size_t)BH * LQK * HD;          // per tensor
  const size_t need  = 2 * elems * sizeof(_Float16);   // Kh + Vh = 16 MB

  if (ws_size >= need) {
    _Float16* Kh = (_Float16*)d_ws;
    _Float16* Vh = Kh + elems;
    eia_cvt_kv<<<dim3(LQK / 64, BH), 256, 0, stream>>>(k, v, Kh, Vh);
    eia_fwd_async<<<dim3(LQK / QT, BH), 256, 0, stream>>>(q, Kh, Vh, kl, out);
  } else {
    eia_fwd_fused<<<dim3(LQK / QT, BH), 256, 0, stream>>>(q, k, v, kl, out);
  }
}
